// xLSTM_37374805409912
// MI455X (gfx1250) — compile-verified, matrix operand order checked by reading, not run on hardware
//
#include <hip/hip_runtime.h>
#include <stdint.h>
#include <stddef.h>

// ---------------- model constants ----------------
#define VOC   50257
#define EDIM  768
#define IDIM  1536       // INNER
#define NH    8
#define DHM_  64
#define DHS_  192
#define QKV_  512
#define BB    2
#define SS    1024
#define ROWS  2048       // B*S
#define EPS_  1e-5f

typedef __bf16 bf16_t;
typedef __attribute__((ext_vector_type(16))) __bf16 v16bf;
typedef __attribute__((ext_vector_type(8)))  float  v8f;

static __device__ inline bf16_t f2bf(float f) {
    union { float f; unsigned u; } a; a.f = f;
    unsigned r = a.u + 0x7FFFu + ((a.u >> 16) & 1u);   // round-to-nearest-even
    union { unsigned short s; bf16_t b; } o; o.s = (unsigned short)(r >> 16);
    return o.b;
}

// ---------------- embedding gather ----------------
__global__ __launch_bounds__(256) void k_embed(const int* __restrict__ ids,
                                               const float* __restrict__ emb,
                                               float* __restrict__ x) {
    int row = blockIdx.x;                    // 0..ROWS-1
    int id  = ids[row];
    const float* src = emb + (size_t)id * EDIM;
    float* dst = x + (size_t)row * EDIM;
    for (int i = threadIdx.x; i < EDIM; i += blockDim.x) dst[i] = src[i];
}

// ---------------- layernorm -> bf16 ----------------
__global__ __launch_bounds__(256) void k_ln_bf16(const float* __restrict__ x,
                                                 const float* __restrict__ g,
                                                 const float* __restrict__ b,
                                                 bf16_t* __restrict__ y) {
    __shared__ float s1[256], s2[256];
    int row = blockIdx.x, t = threadIdx.x;
    const float* xr = x + (size_t)row * EDIM;
    float v0 = xr[t], v1 = xr[t + 256], v2 = xr[t + 512];
    s1[t] = v0 + v1 + v2;
    s2[t] = v0 * v0 + v1 * v1 + v2 * v2;
    __syncthreads();
    for (int off = 128; off; off >>= 1) {
        if (t < off) { s1[t] += s1[t + off]; s2[t] += s2[t + off]; }
        __syncthreads();
    }
    float mean = s1[0] * (1.0f / EDIM);
    float var  = s2[0] * (1.0f / EDIM) - mean * mean;
    float rstd = rsqrtf(var + EPS_);
    bf16_t* yr = y + (size_t)row * EDIM;
    yr[t]       = f2bf((v0 - mean) * rstd * g[t]       + b[t]);
    yr[t + 256] = f2bf((v1 - mean) * rstd * g[t + 256] + b[t + 256]);
    yr[t + 512] = f2bf((v2 - mean) * rstd * g[t + 512] + b[t + 512]);
}

// ---------------- flat f32 -> bf16 ----------------
__global__ __launch_bounds__(256) void k_cvt(const float* __restrict__ in,
                                             bf16_t* __restrict__ out, size_t n) {
    size_t i = (size_t)blockIdx.x * blockDim.x + threadIdx.x;
    size_t stride = (size_t)gridDim.x * blockDim.x;
    for (; i < n; i += stride) out[i] = f2bf(in[i]);
}

// ---------------- transpose-convert: f32 [IN][OUT] -> bf16 [OUT][IN] ----------------
__global__ __launch_bounds__(256) void k_tcvt(const float* __restrict__ W,
                                              bf16_t* __restrict__ Wt,
                                              int IN, int OUT) {
    __shared__ float tile[32][33];
    int i0 = blockIdx.x * 32, o0 = blockIdx.y * 32;
    int tx = threadIdx.x & 31, ty = threadIdx.x >> 5;   // 8 rows of 32
    for (int r = ty; r < 32; r += 8) {
        int i = i0 + r, o = o0 + tx;
        tile[r][tx] = (i < IN && o < OUT) ? W[(size_t)i * OUT + o] : 0.0f;
    }
    __syncthreads();
    for (int r = ty; r < 32; r += 8) {
        int o = o0 + r, i = i0 + tx;
        if (o < OUT && i < IN) Wt[(size_t)o * IN + i] = f2bf(tile[tx][r]);
    }
}

// ---------------- WMMA bf16 GEMM: C[M][N] = A[M][K] * Bt[N][K]^T ----------------
// one wave computes a 32x64 tile of C (8 x v_wmma_f32_16x16x32_bf16 per K step).
// Ragged N handled branch-free: B row index clamped to N-1 (VALU select, no EXEC
// divergence in the K loop -> WMMA always runs with full EXEC), stores guarded.
// Optional fused epilogues: f32 store (Cf) and/or bf16 store (Cb).
__global__ __launch_bounds__(256) void k_gemm(const bf16_t* __restrict__ A,
                                              const bf16_t* __restrict__ Bt,
                                              float* __restrict__ Cf,
                                              bf16_t* __restrict__ Cb,
                                              int M, int Kd, int N) {
    int wave = blockIdx.x * 8 + (threadIdx.x >> 5);
    int lane = threadIdx.x & 31;
    int tilesM = M >> 5;                 // 32-row strips (M always multiple of 32)
    int tilesN = (N + 63) >> 6;          // 64-col strips
    if (wave >= tilesM * tilesN) return; // wave-uniform exit
    int tm = wave % tilesM;
    int tn = wave / tilesM;
    int hf    = lane >> 4;               // lane half selects K sub-runs
    int idx16 = lane & 15;
    const bf16_t* arow0 = A + (size_t)(tm * 32 + idx16) * Kd;
    const bf16_t* arow1 = arow0 + (size_t)16 * Kd;
    int nbase = tn * 64;
    const bf16_t* brow[4];
#pragma unroll
    for (int j = 0; j < 4; ++j) {
        int n = nbase + j * 16 + idx16;
        if (n > N - 1) n = N - 1;        // clamp: junk columns never stored
        brow[j] = Bt + (size_t)n * Kd;
    }
    v8f acc[8] = {};                     // [mi*4 + j]
    for (int kk = 0; kk < Kd; kk += 32) {
        if (kk + 64 < Kd) {
            __builtin_prefetch(arow0 + kk + 64, 0, 1);
            __builtin_prefetch(arow1 + kk + 64, 0, 1);
        }
        union { v16bf v; uint4 u[2]; } fa0, fa1;
        // A 16x32 bf16 layout: lanes 0-15 hold K in [0,8)+[16,24), lanes 16-31 complement
        fa0.u[0] = *(const uint4*)(arow0 + kk + hf * 8);
        fa0.u[1] = *(const uint4*)(arow0 + kk + 16 + hf * 8);
        fa1.u[0] = *(const uint4*)(arow1 + kk + hf * 8);
        fa1.u[1] = *(const uint4*)(arow1 + kk + 16 + hf * 8);
#pragma unroll
        for (int j = 0; j < 4; ++j) {
            union { v16bf v; uint4 u[2]; } fb;
            fb.u[0] = *(const uint4*)(brow[j] + kk + hf * 8);
            fb.u[1] = *(const uint4*)(brow[j] + kk + 16 + hf * 8);
            acc[j] = __builtin_amdgcn_wmma_f32_16x16x32_bf16(
                false, fa0.v, false, fb.v, (short)0, acc[j], false, false);
            acc[4 + j] = __builtin_amdgcn_wmma_f32_16x16x32_bf16(
                false, fa1.v, false, fb.v, (short)0, acc[4 + j], false, false);
        }
    }
    // D layout: VGPR r -> row r (lanes 0-15) / row 8+r (lanes 16-31), col = lane&15
#pragma unroll
    for (int mi = 0; mi < 2; ++mi) {
        int rbase = tm * 32 + mi * 16 + hf * 8;
#pragma unroll
        for (int j = 0; j < 4; ++j) {
            int col = nbase + j * 16 + idx16;
            if (col < N) {
                v8f a = acc[mi * 4 + j];
#pragma unroll
                for (int r = 0; r < 8; ++r) {
                    size_t off = (size_t)(rbase + r) * N + col;
                    if (Cf) Cf[off] = a[r];
                    if (Cb) Cb[off] = f2bf(a[r]);
                }
            }
        }
    }
}

// ---------------- causal depthwise conv (K=4) ----------------
__global__ __launch_bounds__(256) void k_dwconv(const float* __restrict__ h,
                                                const float* __restrict__ w,
                                                const float* __restrict__ bias,
                                                float* __restrict__ y) {
    int bs = blockIdx.x;                 // row = b*SS + s
    int s  = bs & (SS - 1);
    int b  = bs >> 10;
    for (int c = threadIdx.x; c < IDIM; c += blockDim.x) {
        float acc = bias[c];
#pragma unroll
        for (int j = 0; j < 4; ++j) {
            int sp = s - 3 + j;
            if (sp >= 0) acc += w[c * 4 + j] * h[((size_t)b * SS + sp) * IDIM + c];
        }
        y[(size_t)bs * IDIM + c] = acc;
    }
}

// ---------------- mLSTM sequential scan: one block per (b,h), C state in VGPRs ----------------
__global__ __launch_bounds__(64) void k_mlstm(const float* __restrict__ q,
                                              const float* __restrict__ k,
                                              const float* __restrict__ v,
                                              const float* __restrict__ og,
                                              const float* __restrict__ gif,
                                              const float* __restrict__ bif,
                                              float* __restrict__ hout) {
    int b = blockIdx.x >> 3, h = blockIdx.x & 7;
    int d = threadIdx.x;                 // 0..63 : row of C
    float Cr[64];
#pragma unroll
    for (int e = 0; e < 64; ++e) Cr[e] = 0.0f;
    float n_d = 0.0f, m = 0.0f;
    __shared__ float qsh[64], ksh[64], red[64];
    const float kscale = 0.125f;         // 1/sqrt(64)
    for (int s = 0; s < SS; ++s) {
        size_t bs   = (size_t)b * SS + s;
        size_t base = (bs * NH + h) * DHM_;
        float qt = q[base + d];
        float kt = k[base + d] * kscale;
        float vt = v[base + d];
        float ot = og[base + d];
        float it = gif[bs * 16 + h]     + bif[h];
        float ft = gif[bs * 16 + 8 + h] + bif[8 + h];
        qsh[d] = qt; ksh[d] = kt;
        __syncthreads();
        float m_new = fmaxf(ft + m, it);
        float iex = __expf(it - m_new);
        float fex = __expf(ft + m - m_new);
        float iv  = iex * vt;
        float hnum = 0.0f;
#pragma unroll
        for (int e = 0; e < 64; ++e) {
            Cr[e] = fex * Cr[e] + iv * ksh[e];
            hnum += Cr[e] * qsh[e];
        }
        n_d = fex * n_d + iex * kt;
        red[d] = n_d * qt;
        __syncthreads();
        for (int off = 32; off; off >>= 1) {
            if (d < off) red[d] += red[d + off];
            __syncthreads();
        }
        float den = fmaxf(fabsf(red[0]), 1.0f);
        float hv  = (1.0f / (1.0f + __expf(-ot))) * hnum / den;
        hout[base + d] = hv;
        m = m_new;
        __syncthreads();
    }
}

// ---------------- sLSTM sequential scan: one block per (b,head) ----------------
__global__ __launch_bounds__(256) void k_slstm(const float* __restrict__ gx,
                                               const float* __restrict__ R,
                                               const float* __restrict__ bg,
                                               float* __restrict__ hs) {
    int b = blockIdx.x >> 3, hh = blockIdx.x & 7;
    int t = threadIdx.x;
    __shared__ float hsh[DHS_];
    __shared__ float pre[4 * DHS_];
    float c = 0.0f, n = 0.0f, m = 0.0f;
    if (t < DHS_) hsh[t] = 0.0f;
    __syncthreads();
    for (int s = 0; s < SS; ++s) {
        size_t bs = (size_t)b * SS + s;
        size_t gxbase = bs * (4 * IDIM);                 // [4][8][192]
#pragma unroll
        for (int rep = 0; rep < 3; ++rep) {
            int o = t + rep * 256;                        // 0..767
            int g = o / DHS_, kc = o - g * DHS_;
            float acc = gx[gxbase + (size_t)g * IDIM + hh * DHS_ + kc]
                      + bg[(g * NH + hh) * DHS_ + kc];
            const float* Rcol = R + ((size_t)(g * NH + hh) * DHS_) * DHS_ + kc;
            for (int dd = 0; dd < DHS_; ++dd) acc += hsh[dd] * Rcol[(size_t)dd * DHS_];
            pre[o] = acc;
        }
        __syncthreads();
        if (t < DHS_) {
            float it = pre[t], ft = pre[DHS_ + t], zt = pre[2 * DHS_ + t], ot = pre[3 * DHS_ + t];
            float m_new = fmaxf(ft + m, it);
            float i = __expf(it - m_new), f = __expf(ft + m - m_new);
            c = f * c + i * tanhf(zt);
            n = f * n + i;
            float hv = (1.0f / (1.0f + __expf(-ot))) * c / fmaxf(n, 1e-6f);
            hsh[t] = hv;
            hs[bs * IDIM + hh * DHS_ + t] = hv;
            m = m_new;
        }
        __syncthreads();
    }
}

// ---------------- residual add ----------------
__global__ __launch_bounds__(256) void k_add(float* __restrict__ x,
                                             const float* __restrict__ y, size_t n) {
    size_t i = (size_t)blockIdx.x * blockDim.x + threadIdx.x;
    size_t stride = (size_t)gridDim.x * blockDim.x;
    for (; i < n; i += stride) x[i] += y[i];
}

// ================= host orchestration =================
struct MB { const float *ln_g,*ln_b,*up,*down,*Wq,*Wk,*Wv,*Wif,*bif,*Wo,*out; };
struct SB { const float *ln_g,*ln_b,*up,*down,*conv_w,*conv_b,*W,*bg,*R; };

static inline void gemm_launch(const bf16_t* A, const bf16_t* Bt,
                               float* Cf, bf16_t* Cb,
                               int M, int Kd, int N, hipStream_t st) {
    int tiles  = (M / 32) * ((N + 63) / 64);
    int blocks = (tiles + 7) / 8;
    k_gemm<<<blocks, 256, 0, st>>>(A, Bt, Cf, Cb, M, Kd, N);
}
static inline void tcvt_launch(const float* W, bf16_t* Wt, int IN, int OUT, hipStream_t st) {
    dim3 g((IN + 31) / 32, (OUT + 31) / 32);
    k_tcvt<<<g, 256, 0, st>>>(W, Wt, IN, OUT);
}
static inline void cvt_launch(const float* in, bf16_t* out, size_t n, hipStream_t st) {
    int blocks = (int)((n + 255 * 8) / (256 * 8));
    if (blocks < 1) blocks = 1;
    if (blocks > 4096) blocks = 4096;
    k_cvt<<<blocks, 256, 0, st>>>(in, out, n);
}

extern "C" void kernel_launch(void* const* d_in, const int* in_sizes, int n_in,
                              void* d_out, int out_size, void* d_ws, size_t ws_size,
                              hipStream_t stream) {
    const int* ids = (const int*)d_in[0];
    const float *emb = nullptr, *lnf_g = nullptr, *lnf_b = nullptr;
    MB mb[3]; SB sb;
    int idx = 1;
    auto nxt = [&]() { return (const float*)d_in[idx++]; };
    bool insertion = (in_sizes[1] == VOC * EDIM);
    if (insertion) {
        emb = nxt(); lnf_g = nxt(); lnf_b = nxt();
        for (int i = 0; i < 3; ++i) {
            mb[i].ln_g = nxt(); mb[i].ln_b = nxt(); mb[i].up = nxt(); mb[i].down = nxt();
            mb[i].Wq = nxt(); mb[i].Wk = nxt(); mb[i].Wv = nxt(); mb[i].Wif = nxt();
            mb[i].bif = nxt(); mb[i].Wo = nxt(); mb[i].out = nxt();
        }
        sb.ln_g = nxt(); sb.ln_b = nxt(); sb.up = nxt(); sb.down = nxt();
        sb.conv_w = nxt(); sb.conv_b = nxt(); sb.W = nxt(); sb.bg = nxt(); sb.R = nxt();
    } else {  // JAX sorted-key tree flatten: blocks first, then emb, lnf_b, lnf_g
        for (int i = 0; i < 3; ++i) {
            mb[i].Wif = nxt(); mb[i].Wk = nxt(); mb[i].Wo = nxt(); mb[i].Wq = nxt();
            mb[i].Wv = nxt(); mb[i].bif = nxt(); mb[i].down = nxt();
            mb[i].ln_b = nxt(); mb[i].ln_g = nxt(); mb[i].out = nxt(); mb[i].up = nxt();
        }
        sb.R = nxt(); sb.W = nxt(); sb.bg = nxt(); sb.conv_b = nxt(); sb.conv_w = nxt();
        sb.down = nxt(); sb.ln_b = nxt(); sb.ln_g = nxt(); sb.up = nxt();
        emb = nxt(); lnf_b = nxt(); lnf_g = nxt();
    }

    // -------- workspace carve --------
    char* wsb = (char*)d_ws; size_t cur = 0;
    auto carve = [&](size_t bytes) -> void* {
        void* p = wsb + cur;
        cur = (cur + bytes + 255) & ~(size_t)255;
        return p;
    };
    float*  x      = (float*) carve((size_t)ROWS * EDIM * 4);    // residual stream
    bf16_t* aE     = (bf16_t*)carve((size_t)ROWS * EDIM * 2);    // bf16 A, K=768
    bf16_t* aI     = (bf16_t*)carve((size_t)ROWS * IDIM * 2);    // bf16 A, K=1536
    bf16_t* aQ     = (bf16_t*)carve((size_t)ROWS * QKV_ * 2);    // bf16 A, K=512
    float*  hup    = (float*) carve((size_t)ROWS * IDIM * 4);    // s-block conv input
    float*  qb     = (float*) carve((size_t)ROWS * QKV_ * 4);
    float*  kb     = (float*) carve((size_t)ROWS * QKV_ * 4);
    float*  vb     = (float*) carve((size_t)ROWS * QKV_ * 4);
    float*  ob     = (float*) carve((size_t)ROWS * QKV_ * 4);
    float*  gifb   = (float*) carve((size_t)ROWS * 16 * 4);
    float*  hatt   = (float*) carve((size_t)ROWS * QKV_ * 4);
    float*  tmpI   = (float*) carve((size_t)ROWS * IDIM * 4);    // conv_out / hs
    float*  gxb    = (float*) carve((size_t)ROWS * 4 * IDIM * 4);
    float*  dbuf   = (float*) carve((size_t)ROWS * EDIM * 4);
    bf16_t* wbuf   = (bf16_t*)carve((size_t)VOC * EDIM * 2);     // largest: emb in bf16

    // -------- forward --------
    k_embed<<<ROWS, 256, 0, stream>>>(ids, emb, x);

    for (int blk = 0; blk < 4; ++blk) {
        const bool is_s = (blk == 3);
        const float *ln_g = is_s ? sb.ln_g : mb[blk].ln_g;
        const float *ln_b = is_s ? sb.ln_b : mb[blk].ln_b;
        const float *up   = is_s ? sb.up   : mb[blk].up;
        const float *down = is_s ? sb.down : mb[blk].down;

        k_ln_bf16<<<ROWS, 256, 0, stream>>>(x, ln_g, ln_b, aE);
        tcvt_launch(up, wbuf, EDIM, IDIM, stream);
        if (!is_s) {
            // mLSTM: up-proj consumed only by downstream GEMMs -> bf16 only
            gemm_launch(aE, wbuf, nullptr, aI, ROWS, EDIM, IDIM, stream);

            tcvt_launch(mb[blk].Wq, wbuf, IDIM, QKV_, stream);
            gemm_launch(aI, wbuf, qb, nullptr, ROWS, IDIM, QKV_, stream);
            tcvt_launch(mb[blk].Wk, wbuf, IDIM, QKV_, stream);
            gemm_launch(aI, wbuf, kb, nullptr, ROWS, IDIM, QKV_, stream);
            tcvt_launch(mb[blk].Wv, wbuf, IDIM, QKV_, stream);
            gemm_launch(aI, wbuf, vb, nullptr, ROWS, IDIM, QKV_, stream);
            tcvt_launch(mb[blk].Wo, wbuf, IDIM, QKV_, stream);
            gemm_launch(aI, wbuf, ob, nullptr, ROWS, IDIM, QKV_, stream);
            tcvt_launch(mb[blk].Wif, wbuf, IDIM, 16, stream);
            gemm_launch(aI, wbuf, gifb, nullptr, ROWS, IDIM, 16, stream);

            k_mlstm<<<BB * NH, 64, 0, stream>>>(qb, kb, vb, ob, gifb, mb[blk].bif, hatt);

            cvt_launch(hatt, aQ, (size_t)ROWS * QKV_, stream);
            tcvt_launch(mb[blk].out, wbuf, QKV_, IDIM, stream);
            // attn-out consumed only by down-proj GEMM -> bf16 only (fused epilogue)
            gemm_launch(aQ, wbuf, nullptr, aI, ROWS, QKV_, IDIM, stream);
        } else {
            // sLSTM: conv needs f32 up-proj
            gemm_launch(aE, wbuf, hup, nullptr, ROWS, EDIM, IDIM, stream);
            k_dwconv<<<ROWS, 256, 0, stream>>>(hup, sb.conv_w, sb.conv_b, tmpI);
            cvt_launch(tmpI, aI, (size_t)ROWS * IDIM, stream);
            tcvt_launch(sb.W, wbuf, IDIM, 4 * IDIM, stream);
            gemm_launch(aI, wbuf, gxb, nullptr, ROWS, IDIM, 4 * IDIM, stream);

            k_slstm<<<BB * NH, 256, 0, stream>>>(gxb, sb.R, sb.bg, tmpI);
            cvt_launch(tmpI, aI, (size_t)ROWS * IDIM, stream);
        }

        tcvt_launch(down, wbuf, IDIM, EDIM, stream);
        gemm_launch(aI, wbuf, dbuf, nullptr, ROWS, IDIM, EDIM, stream);
        k_add<<<2048, 256, 0, stream>>>(x, dbuf, (size_t)ROWS * EDIM);
    }

    // final LN + unembedding:  logits = lnf(x) @ emb^T   (Bt == emb[V][E] directly)
    k_ln_bf16<<<ROWS, 256, 0, stream>>>(x, lnf_g, lnf_b, aE);
    cvt_launch(emb, wbuf, (size_t)VOC * EDIM, stream);
    gemm_launch(aE, wbuf, (float*)d_out, nullptr, ROWS, EDIM, VOC, stream);
}
